// PolicyBasedUtteranceRationalExtractor_47278999994943
// MI455X (gfx1250) — matrix operand order repaired and checked by reading, not compiled, not run on hardware
//
#include <hip/hip_runtime.h>
#include <hip/hip_bf16.h>

typedef __attribute__((ext_vector_type(16))) _Float16 v16h;
typedef __attribute__((ext_vector_type(8)))  float    v8f;
typedef __attribute__((ext_vector_type(4)))  float    v4f;
typedef __attribute__((ext_vector_type(4)))  unsigned int u32x4;
typedef __attribute__((ext_vector_type(8)))  int      i32x8;
typedef __attribute__((ext_vector_type(4)))  int      i32x4;

#define N_UTT   128
#define NB      64          // batch
#define E_DIM   256
#define H_DIM   128
#define G4H     512         // 4*H
#define M_ALL   (N_UTT * NB)   // 8192
#define L_SEQ   2048

// ---- workspace layout (bytes) ----
constexpr size_t OFF_UH   = 0;                                  // u f16 [8192][256]
constexpr size_t OFF_WIH  = OFF_UH   + (size_t)M_ALL * E_DIM * 2;   // w_ih f16 [2][512][256]
constexpr size_t OFF_WHH  = OFF_WIH  + 2ull * G4H * E_DIM * 2;      // w_hh f16 [2][512][128]
constexpr size_t OFF_BIAS = OFF_WHH  + 2ull * G4H * H_DIM * 2;      // bias f32 [2][512]
constexpr size_t OFF_XP   = OFF_BIAS + 2ull * G4H * 4;              // xp  f32 [2][8192][512]
constexpr size_t OFF_H    = OFF_XP   + 2ull * M_ALL * G4H * 4;      // h   f32 [2][8192][128]

// ---- output layout (f32 elements) ----
constexpr int OUT_LOGITS = 0;       // [128][64][2]
constexpr int OUT_POLICY = 16384;   // [128][64][2]
constexpr int OUT_CHOSEN = 32768;   // [128][64][1]
constexpr int OUT_MASK   = 40960;   // [128][64]
constexpr int OUT_MASKED = 49152;   // [2048][64]

// ---- LDS strides for scan kernel (padded against bank conflicts) ----
#define GLD 516   // G f32 row stride
#define HLD 136   // hA f16 row stride
#define WLD 136   // w_hh LDS f16 row stride (matches TDM pad: 64 dwords + 4 dwords pad)
constexpr int SMEM_G   = NB * GLD * 4;        // 132096
constexpr int SMEM_HA  = NB * HLD * 2;        // 17408
constexpr int SMEM_WHH = G4H * WLD * 2;       // 139264
constexpr int SMEM_SCAN = SMEM_G + SMEM_HA + SMEM_WHH;   // 288768 <= 320KB/WGP

union Frag32B { v16h v; v4f q[2]; };

// A fragment 16x32 f16 (ISA 7.12.2): lanes 0-15 rows 0-15 K{0..7,16..23};
// lanes 16-31 same rows, K{8..15,24..31}. Source row-major [M][K], ld elems.
__device__ __forceinline__ v16h load_a_frag(const _Float16* p, int ld) {
    int lane = threadIdx.x & 31;
    int row  = lane & 15;
    int kk   = (lane & 16) >> 1;           // 0 or 8
    const _Float16* q = p + row * ld + kk;
    Frag32B f;
    f.q[0] = *(const v4f*)(q);             // K = kk .. kk+7
    f.q[1] = *(const v4f*)(q + 16);        // K = kk+16 .. kk+23
    return f.v;
}

// B fragment 32x16 f16: lane n(0-15) holds col n, K{0..7,16..23}; lanes 16-31 K{8..15,24..31}.
// Source row-major w[N][K] (natural storage of w_ih / w_hh), ld elems.
__device__ __forceinline__ v16h load_b_frag(const _Float16* p, int ld) {
    int lane = threadIdx.x & 31;
    int col  = lane & 15;
    int kk   = (lane & 16) >> 1;           // 0 or 8
    const _Float16* q = p + col * ld + kk;
    Frag32B f;
    f.q[0] = *(const v4f*)(q);
    f.q[1] = *(const v4f*)(q + 16);
    return f.v;
}

__device__ __forceinline__ float sigmoidf(float x) {
    return 1.0f / (1.0f + __expf(-x));
}

// ---------------- kernel 0a: f32 -> f16 copy ----------------
__global__ void k_f32_to_f16(const float* __restrict__ src, _Float16* __restrict__ dst, int n) {
    int i = blockIdx.x * blockDim.x + threadIdx.x;
    if (i < n) dst[i] = (_Float16)src[i];
}

// ---------------- kernel 0b: bias combine ----------------
__global__ void k_bias_add(const float* __restrict__ a, const float* __restrict__ b,
                           float* __restrict__ o, int n) {
    int i = blockIdx.x * blockDim.x + threadIdx.x;
    if (i < n) o[i] = a[i] + b[i];
}

// ---------------- kernel 1: embedding gather + mean pool -> u f16 ----------------
__global__ void k_gather_pool(const int* __restrict__ x, const float* __restrict__ emb,
                              _Float16* __restrict__ u_h) {
    int tb = blockIdx.x;               // t*64 + b
    int t = tb >> 6, b = tb & 63;
    __shared__ int toks[16];
    if (threadIdx.x < 16) toks[threadIdx.x] = x[(t * 16 + threadIdx.x) * NB + b];
    __syncthreads();
    int e = threadIdx.x;
    float s = 0.0f;
#pragma unroll
    for (int j = 0; j < 16; ++j) s += emb[(size_t)toks[j] * E_DIM + e];
    u_h[(size_t)tb * E_DIM + e] = (_Float16)(s * 0.0625f);   // every utterance has 16 tokens
}

// ---------------- kernel 2: xp = u @ w_ih^T + (b_ih + b_hh), both dirs ----------------
// one wave per 16x64 tile; K=256, double-buffered 32-wide k-chunks so global
// loads overlap the WMMAs instead of wait-to-zero before every matrix op.
__global__ void k_xp_gemm(const _Float16* __restrict__ u, const _Float16* __restrict__ wih,
                          const float* __restrict__ bias, float* __restrict__ xp) {
    int wave = (blockIdx.x * blockDim.x + threadIdx.x) >> 5;   // 0..8191
    int dir  = wave >> 12;
    int rem  = wave & 4095;
    int m0   = (rem >> 3) << 4;        // 512 m-tiles
    int n0   = (rem & 7) << 6;         // 8 groups of 64 cols
    const _Float16* A  = u   + (size_t)m0 * E_DIM;
    const _Float16* W  = wih + (size_t)dir * G4H * E_DIM;
    const float*    bs = bias + dir * G4H;
    float*          out = xp + (size_t)dir * M_ALL * G4H;

    v8f z = {0.f,0.f,0.f,0.f,0.f,0.f,0.f,0.f};
    v8f acc[4] = {z, z, z, z};

    v16h a_cur = load_a_frag(A, E_DIM);
    v16h b_cur[4];
#pragma unroll
    for (int s = 0; s < 4; ++s)
        b_cur[s] = load_b_frag(W + (size_t)(n0 + s * 16) * E_DIM, E_DIM);

#pragma unroll
    for (int k0 = 0; k0 < E_DIM; k0 += 32) {
        v16h a_nxt;
        v16h b_nxt[4];
        if (k0 + 32 < E_DIM) {                 // prefetch next k-chunk first
            a_nxt = load_a_frag(A + k0 + 32, E_DIM);
#pragma unroll
            for (int s = 0; s < 4; ++s)
                b_nxt[s] = load_b_frag(W + (size_t)(n0 + s * 16) * E_DIM + k0 + 32, E_DIM);
        }
#pragma unroll
        for (int s = 0; s < 4; ++s)
            acc[s] = __builtin_amdgcn_wmma_f32_16x16x32_f16(
                false, a_cur, false, b_cur[s], (short)0, acc[s], false, false);
        if (k0 + 32 < E_DIM) {
            a_cur = a_nxt;
#pragma unroll
            for (int s = 0; s < 4; ++s) b_cur[s] = b_nxt[s];
        }
    }

    int lane = threadIdx.x & 31;
    int col  = lane & 15;
    int r0   = (lane & 16) >> 1;       // 0 or 8
#pragma unroll
    for (int s = 0; s < 4; ++s) {
        int n = n0 + s * 16 + col;
        float bv = bs[n];
#pragma unroll
        for (int r = 0; r < 8; ++r)
            out[(size_t)(m0 + r0 + r) * G4H + n] = acc[s][r] + bv;
    }
}

// ---------------- kernel 3: persistent BiLSTM scan (one block per direction) ----------------
// 512 threads = 16 waves. Per step: G[64][512] = h@w_hh^T + xp[t] via 512 WMMAs,
// then gates; h (f16) and full w_hh (f16, TDM-staged with LDS padding) live in
// LDS; c stays in registers with a fixed thread<->element mapping.
__global__ void __launch_bounds__(512) k_bilstm_scan(const _Float16* __restrict__ whh_g,
                                                     const float* __restrict__ xp,
                                                     float* __restrict__ hout) {
    int dir = blockIdx.x;
    const _Float16* Wg = whh_g + (size_t)dir * G4H * H_DIM;
    const float*    XP = xp    + (size_t)dir * M_ALL * G4H;
    float*          HO = hout  + (size_t)dir * M_ALL * H_DIM;

    extern __shared__ char smem[];
    float*    G  = (float*)smem;                          // [64][GLD]
    _Float16* hA = (_Float16*)(smem + SMEM_G);            // [64][HLD]
    _Float16* Wl = (_Float16*)(smem + SMEM_G + SMEM_HA);  // [512][WLD]

    int tid = threadIdx.x;

#if __has_builtin(__builtin_amdgcn_tensor_load_to_lds) && __has_builtin(__builtin_amdgcn_s_wait_tensorcnt)
    // ---- Tensor Data Mover: stage w_hh [512 rows x 128 f16] into LDS with
    // padding (64 dwords data + 4 dwords pad per row -> WLD=136 halves). ----
    if (tid < 32) {
        unsigned long long ga = (unsigned long long)(uintptr_t)Wg;
        unsigned lds_off = (unsigned)(uintptr_t)Wl;       // low 32 bits = LDS byte addr
        u32x4 g0;
        g0.x = 1u;                                        // count=1, user mode
        g0.y = lds_off;                                   // lds_addr [63:32]
        g0.z = (unsigned)(ga & 0xFFFFFFFFu);              // global_addr [95:64]
        g0.w = (unsigned)((ga >> 32) & 0x01FFFFFFu)       // global_addr [120:96]
             | (2u << 30);                                // type=2 ("image") [127:126]
        i32x8 g1;
        g1[0] = (1 << 16)        // data_size = 1 -> 2 bytes
              | (1 << 20)        // pad_enable
              | (5 << 22)        // pad_interval: 64 dwords (one 128-half row)
              | (3 << 25);       // pad_amount: 4 dwords (8 halves)
        g1[1] = (int)(128u << 16);   // tensor_dim0[15:0] = 128 elems/row
        g1[2] = (int)(512u << 16);   // tensor_dim1[15:0] = 512 rows
        g1[3] = (int)(128u << 16);   // tile_dim0 = 128
        g1[4] = 512;                 // tile_dim1 = 512
        g1[5] = 128;                 // tensor_dim0_stride = 128 (rows contiguous)
        g1[6] = 0;
        g1[7] = 0;
        i32x4 zz4 = {0, 0, 0, 0};
        i32x8 zz8 = {0, 0, 0, 0, 0, 0, 0, 0};
        __builtin_amdgcn_tensor_load_to_lds(g0, g1, zz4, zz4, zz8, 0);
        __builtin_amdgcn_s_wait_tensorcnt((short)0);
    }
#else
    {
        const uint4* src = (const uint4*)(Wg + (size_t)tid * H_DIM);
        uint4*       dst = (uint4*)(Wl + (size_t)tid * WLD);
#pragma unroll
        for (int q = 0; q < 16; ++q) dst[q] = src[q];
    }
#endif

    // zero h
    for (int i = tid; i < NB * HLD; i += 512) hA[i] = (_Float16)0.0f;

    float c_reg[16];
#pragma unroll
    for (int k = 0; k < 16; ++k) c_reg[k] = 0.0f;

    __syncthreads();

    int wave = tid >> 5;
    int lane = tid & 31;
    int col  = lane & 15;
    int r0   = (lane & 16) >> 1;           // 0 or 8
    int m0   = (wave >> 2) << 4;           // all 8 tiles of a wave share m0
    int nbase = (wave & 3) << 3;           // tile i -> n-tile nbase+i

    for (int s = 0; s < 128; ++s) {
        int t = dir ? (127 - s) : s;
        const float* xpt = XP + (size_t)t * NB * G4H;
        if (s + 1 < 128) {                 // prefetch next step's gate pre-activations
            int tn = dir ? (126 - s) : (s + 1);
            __builtin_prefetch(XP + (size_t)tn * NB * G4H + tid * 64, 0, 1);
        }
        // A fragments: loaded once per step, reused by all 8 n-tiles of this wave
        v16h a[4];
#pragma unroll
        for (int k = 0; k < 4; ++k)
            a[k] = load_a_frag(hA + m0 * HLD + 32 * k, HLD);

#pragma unroll
        for (int i = 0; i < 8; ++i) {
            int n0 = (nbase + i) << 4;
            v8f acc;
#pragma unroll
            for (int r = 0; r < 8; ++r)
                acc[r] = xpt[(size_t)(m0 + r0 + r) * G4H + n0 + col];
#pragma unroll
            for (int k = 0; k < 4; ++k) {
                v16h b = load_b_frag(Wl + n0 * WLD + 32 * k, WLD);
                acc = __builtin_amdgcn_wmma_f32_16x16x32_f16(
                    false, a[k], false, b, (short)0, acc, false, false);
            }
#pragma unroll
            for (int r = 0; r < 8; ++r)
                G[(m0 + r0 + r) * GLD + n0 + col] = acc[r];
        }
        __syncthreads();
        // --- gates: 16 fixed elements per thread, c in registers ---
#pragma unroll
        for (int k = 0; k < 16; ++k) {
            int e = tid + (k << 9);
            int m = e >> 7, j = e & 127;
            float gi = G[m * GLD + j];
            float gf = G[m * GLD + H_DIM + j];
            float gg = G[m * GLD + 2 * H_DIM + j];
            float go = G[m * GLD + 3 * H_DIM + j];
            float cc = sigmoidf(gf) * c_reg[k] + sigmoidf(gi) * tanhf(gg);
            float hh = sigmoidf(go) * tanhf(cc);
            c_reg[k] = cc;
            HO[((size_t)t * NB + m) * H_DIM + j] = hh;
            hA[m * HLD + j] = (_Float16)hh;
        }
        __syncthreads();
    }
}

// ---------------- kernel 4: head: logits / softmax / argmax / mask expansion ----------------
__global__ void k_head(const float* __restrict__ hf, const float* __restrict__ hb,
                       const float* __restrict__ w_out, const float* __restrict__ b_out,
                       const int* __restrict__ x, float* __restrict__ out) {
    int id = blockIdx.x * blockDim.x + threadIdx.x;   // 0..8191 = t*64+b
    int t = id >> 6, b = id & 63;
    const float* pf = hf + (size_t)id * H_DIM;
    const float* pb = hb + (size_t)id * H_DIM;
    float l0 = b_out[0], l1 = b_out[1];
#pragma unroll 4
    for (int e = 0; e < H_DIM; ++e) {
        float v = pf[e];
        l0 += w_out[e] * v;                   // w_out[0][0:128]
        l1 += w_out[E_DIM + e] * v;           // w_out[1][0:128]
    }
#pragma unroll 4
    for (int e = 0; e < H_DIM; ++e) {
        float v = pb[e];
        l0 += w_out[H_DIM + e] * v;           // w_out[0][128:256]
        l1 += w_out[E_DIM + H_DIM + e] * v;   // w_out[1][128:256]
    }
    float mx = fmaxf(l0, l1);
    float e0 = __expf(l0 - mx), e1 = __expf(l1 - mx);
    float inv = 1.0f / (e0 + e1);
    float p0 = e0 * inv, p1 = e1 * inv;
    int am = (p1 > p0) ? 1 : 0;               // argmax, first-max ties -> 0
    out[OUT_LOGITS + id * 2 + 0] = l0;
    out[OUT_LOGITS + id * 2 + 1] = l1;
    out[OUT_POLICY + id * 2 + 0] = p0;
    out[OUT_POLICY + id * 2 + 1] = p1;
    out[OUT_CHOSEN + id] = am ? p1 : p0;
    out[OUT_MASK + id] = (float)am;
#pragma unroll
    for (int j = 0; j < 16; ++j) {
        int l = t * 16 + j;
        out[OUT_MASKED + l * NB + b] = am ? (float)x[l * NB + b] : 0.0f;
    }
}

extern "C" void kernel_launch(void* const* d_in, const int* in_sizes, int n_in,
                              void* d_out, int out_size, void* d_ws, size_t ws_size,
                              hipStream_t stream) {
    (void)in_sizes; (void)n_in; (void)out_size; (void)ws_size;
    const int*   x      = (const int*)d_in[0];
    const float* emb    = (const float*)d_in[1];
    const float* w_ih_f = (const float*)d_in[2];
    const float* w_hh_f = (const float*)d_in[3];
    const float* b_ih_f = (const float*)d_in[4];
    const float* b_hh_f = (const float*)d_in[5];
    const float* w_ih_b = (const float*)d_in[6];
    const float* w_hh_b = (const float*)d_in[7];
    const float* b_ih_b = (const float*)d_in[8];
    const float* b_hh_b = (const float*)d_in[9];
    const float* w_out  = (const float*)d_in[10];
    const float* b_out  = (const float*)d_in[11];
    float* out = (float*)d_out;

    char* ws = (char*)d_ws;
    _Float16* u_h   = (_Float16*)(ws + OFF_UH);
    _Float16* wih_h = (_Float16*)(ws + OFF_WIH);
    _Float16* whh_h = (_Float16*)(ws + OFF_WHH);
    float*    bias  = (float*)(ws + OFF_BIAS);
    float*    xp    = (float*)(ws + OFF_XP);
    float*    h_ws  = (float*)(ws + OFF_H);

    // 0: weight precision conversion + bias combine
    k_f32_to_f16<<<(G4H * E_DIM + 255) / 256, 256, 0, stream>>>(w_ih_f, wih_h, G4H * E_DIM);
    k_f32_to_f16<<<(G4H * E_DIM + 255) / 256, 256, 0, stream>>>(w_ih_b, wih_h + G4H * E_DIM, G4H * E_DIM);
    k_f32_to_f16<<<(G4H * H_DIM + 255) / 256, 256, 0, stream>>>(w_hh_f, whh_h, G4H * H_DIM);
    k_f32_to_f16<<<(G4H * H_DIM + 255) / 256, 256, 0, stream>>>(w_hh_b, whh_h + G4H * H_DIM, G4H * H_DIM);
    k_bias_add<<<(G4H + 255) / 256, 256, 0, stream>>>(b_ih_f, b_hh_f, bias, G4H);
    k_bias_add<<<(G4H + 255) / 256, 256, 0, stream>>>(b_ih_b, b_hh_b, bias + G4H, G4H);

    // 1: embedding gather + utterance mean-pool (L2-resident table)
    k_gather_pool<<<M_ALL, 256, 0, stream>>>(x, emb, u_h);

    // 2: input projection GEMM, both directions (WMMA f16->f32, double-buffered K)
    k_xp_gemm<<<1024, 256, 0, stream>>>(u_h, wih_h, bias, xp);

    // 3: persistent BiLSTM scan, one workgroup per direction, w_hh/h in LDS
    (void)hipFuncSetAttribute((const void*)k_bilstm_scan,
                              hipFuncAttributeMaxDynamicSharedMemorySize, SMEM_SCAN);
    k_bilstm_scan<<<2, 512, SMEM_SCAN, stream>>>(whh_h, xp, h_ws);

    // 4: output head
    k_head<<<M_ALL / 256, 256, 0, stream>>>(h_ws, h_ws + (size_t)M_ALL * H_DIM,
                                            w_out, b_out, x, out);
}
